// DeepSeekMoE_43276090474633
// MI455X (gfx1250) — matrix operand order repaired and checked by reading, not compiled, not run on hardware
//
#include <hip/hip_runtime.h>

typedef __attribute__((ext_vector_type(8)))  float        v8f;
typedef __attribute__((ext_vector_type(16))) __bf16       v16bf;
typedef __attribute__((ext_vector_type(4)))  unsigned int v4u;
typedef __attribute__((ext_vector_type(8)))  unsigned int v8u;

union ABu {
  v8u   u8;
  v4u   u4[2];
  v16bf bf;
};

#define NTOK   8192
#define DMODEL 1024
#define HDIM   512
#define NEXP   8
#define MTILE  128   // block M tile
#define NTILE  256   // block N tile

__device__ __forceinline__ unsigned short f2bf(float f) {
  unsigned int u = __float_as_uint(f);
  u += 0x7FFFu + ((u >> 16) & 1u);       // round-to-nearest-even
  return (unsigned short)(u >> 16);
}

// ---------------------------------------------------------------- routing ---
__global__ void zero8(int* cnt) {
  if (threadIdx.x < NEXP) cnt[threadIdx.x] = 0;
}

__global__ void scan8(const int* __restrict__ cnt, int* __restrict__ basep) {
  if (threadIdx.x == 0) {
    int s = 0;
    for (int e = 0; e < NEXP; ++e) { basep[e] = s; s += cnt[e]; }
  }
}

__global__ __launch_bounds__(256)
void routing_kernel(const float* __restrict__ X, const float* __restrict__ Wr,
                    const float* __restrict__ eb,
                    int* __restrict__ cnt, int* __restrict__ tok,
                    float* __restrict__ wgt, int N, int d) {
  const int wave = threadIdx.x >> 5;
  const int lane = threadIdx.x & 31;
  const int n = blockIdx.x * 8 + wave;
  if (n >= N) return;
  const float* x = X + (long)n * d;
  float acc[NEXP];
#pragma unroll
  for (int e = 0; e < NEXP; ++e) acc[e] = 0.f;
  for (int k = lane; k < d; k += 32) {
    const float xv = x[k];
#pragma unroll
    for (int e = 0; e < NEXP; ++e) acc[e] += xv * Wr[e * d + k];
  }
#pragma unroll
  for (int e = 0; e < NEXP; ++e)
    for (int m = 16; m >= 1; m >>= 1)
      acc[e] += __shfl_xor(acc[e], m, 32);
  if (lane == 0) {
    float a[NEXP];
#pragma unroll
    for (int e = 0; e < NEXP; ++e) {
      const float l = acc[e] + eb[e];
      const float sp = (l > 15.f) ? l : log1pf(__expf(l));  // softplus
      a[e] = sqrtf(sp);
    }
    int i0 = 0;
#pragma unroll
    for (int e = 1; e < NEXP; ++e) if (a[e] > a[i0]) i0 = e;
    int i1 = (i0 == 0) ? 1 : 0;
#pragma unroll
    for (int e = 0; e < NEXP; ++e) if (e != i0 && a[e] > a[i1]) i1 = e;
    int s0 = atomicAdd(&cnt[i0], 1);
    tok[i0 * NTOK + s0] = n; wgt[i0 * NTOK + s0] = a[i0];
    int s1 = atomicAdd(&cnt[i1], 1);
    tok[i1 * NTOK + s1] = n; wgt[i1 * NTOK + s1] = a[i1];
  }
}

// ---------------------------------------------------------- data movement ---
__global__ void cvt_bf16(const float* __restrict__ src,
                         unsigned short* __restrict__ dst, long n) {
  long i = ((long)blockIdx.x * blockDim.x + threadIdx.x) * 4;
  if (i + 3 < n) {
    const float4 v = *(const float4*)(src + i);
    ushort4 o;
    o.x = f2bf(v.x); o.y = f2bf(v.y); o.z = f2bf(v.z); o.w = f2bf(v.w);
    *(ushort4*)(dst + i) = o;
  } else {
    for (; i < n; ++i) dst[i] = f2bf(src[i]);
  }
}

// src: fp32 [R,C] row-major (batched on z) -> dst: bf16 [C,R] row-major
__global__ __launch_bounds__(256)
void transpose_cvt(const float* __restrict__ src,
                   unsigned short* __restrict__ dst, int R, int C) {
  __shared__ float tile[32][33];
  src += (long)blockIdx.z * R * C;
  dst += (long)blockIdx.z * R * C;
  const int c  = blockIdx.x * 32 + threadIdx.x;
  const int r0 = blockIdx.y * 32;
#pragma unroll
  for (int j = 0; j < 32; j += 8)
    tile[threadIdx.y + j][threadIdx.x] = src[(long)(r0 + threadIdx.y + j) * C + c];
  __syncthreads();
  const int outR = blockIdx.x * 32 + threadIdx.y;
  const int outC = r0 + threadIdx.x;
#pragma unroll
  for (int j = 0; j < 32; j += 8)
    dst[(long)(outR + j) * R + outC] = f2bf(tile[threadIdx.x][threadIdx.y + j]);
}

// ------------------------------------------------------------- WMMA GEMM ---
// Block: 256 threads = 8 waves. Block tile 128(M) x 256(N); waves arranged
// 2(M) x 4(N); each wave owns a 64x64 macro-tile = 4 A-frags x 4 B-frags ->
// 16 v_wmma_f32_16x16x32_bf16 per 32-wide K step.
// A: bf16 [rows, K] row-major.  BT: bf16 [Ncols, K] row-major (pre-transposed).
// EPI 0: H = SiLU(A@B + bias) -> bf16 store at row (base[e]+slot)
// EPI 1: Out = A@B + bias     -> f32 plain store (shared path)
// EPI 2: Out[tok[slot]] += wgt[slot]*(A@B + bias)  -> f32 atomicAdd (routed)
template<int EPI, bool ROUTED>
__global__ __launch_bounds__(256)
void moe_gemm(const unsigned short* __restrict__ A,
              const unsigned short* __restrict__ BT, long bStride,
              const float* __restrict__ bias,
              int K, int Ncols, int Mfixed,
              const int* __restrict__ cnt,
              const int* __restrict__ tokAll,
              const float* __restrict__ wgtAll,
              const int* __restrict__ basep,
              void* __restrict__ Outv) {
  const int e = blockIdx.z;
  int M;
  if constexpr (ROUTED) M = cnt[e]; else M = Mfixed;
  const int rBlock = blockIdx.y * MTILE;
  if (rBlock >= M) return;

  const unsigned short* Bp = BT + (long)e * bStride;
  const float* biasp = bias + (long)e * (ROUTED ? Ncols : 0);
  const int*   tokp  = ROUTED ? (tokAll + e * NTOK) : nullptr;
  const float* wgtp  = ROUTED ? (wgtAll + e * NTOK) : nullptr;
  int base = 0;
  if constexpr (ROUTED) base = basep[e];

  const int lane = threadIdx.x & 31;
  const int w    = threadIdx.x >> 5;
  const int half = lane >> 4;           // 0: lanes 0-15, 1: lanes 16-31
  const int mrow = lane & 15;

  const int rBase = rBlock + (w >> 2) * 64;            // 2 M groups of 64
  const int cBase = blockIdx.x * NTILE + (w & 3) * 64; // 4 N groups of 64

  // A-fragment row pointers (gathered for routed GEMM1, base-offset for GEMM2)
  const unsigned short* ap[4];
#pragma unroll
  for (int u = 0; u < 4; ++u) {
    const int slotA = rBase + u * 16 + mrow;
    long arow;
    if constexpr (EPI == 0 && ROUTED)  arow = (slotA < M) ? (long)tokp[slotA] : 0;
    else if constexpr (EPI == 2)       arow = (long)base + ((slotA < M) ? slotA : 0);
    else                               arow = (slotA < M) ? (long)slotA : 0;
    // A k order used here (contiguous, same as B): lanes<16 hold K 0..15; lanes>=16 hold K 16..31
    ap[u] = A + arow * (long)K + half * 16;
  }
  // B column pointers: lanes<16 hold K 0..15, lanes>=16 hold K 16..31; N=lane&15
  const unsigned short* bp[4];
#pragma unroll
  for (int v = 0; v < 4; ++v)
    bp[v] = Bp + (long)(cBase + v * 16 + mrow) * K + half * 16;

  v8f acc[4][4];
#pragma unroll
  for (int u = 0; u < 4; ++u)
#pragma unroll
    for (int v = 0; v < 4; ++v)
#pragma unroll
      for (int i = 0; i < 8; ++i) acc[u][v][i] = 0.f;

#pragma unroll 1
  for (int k0 = 0; k0 < K; k0 += 32) {
    ABu a[4], b[4];
#pragma unroll
    for (int u = 0; u < 4; ++u) {
      a[u].u4[0] = *(const v4u*)(ap[u] + k0);
      a[u].u4[1] = *(const v4u*)(ap[u] + k0 + 8);
    }
#pragma unroll
    for (int v = 0; v < 4; ++v)
      b[v].u8 = *(const v8u*)(bp[v] + k0);
#pragma unroll
    for (int u = 0; u < 4; ++u)
#pragma unroll
      for (int v = 0; v < 4; ++v)
        acc[u][v] = __builtin_amdgcn_wmma_f32_16x16x32_bf16(
            false, a[u].bf, false, b[v].bf, (short)0, acc[u][v], false, false);
  }

  // Epilogue. C/D layout: col = lane&15 (+16*v), row = VGPR r + 8*(lane>=16)
#pragma unroll
  for (int u = 0; u < 4; ++u) {
    int orow[8]; float scl[8];
    if constexpr (EPI == 2) {
#pragma unroll
      for (int r = 0; r < 8; ++r) {
        const int slot = rBase + u * 16 + half * 8 + r;
        orow[r] = (slot < M) ? tokp[slot] : 0;
        scl[r]  = (slot < M) ? wgtp[slot] : 0.f;
      }
    }
#pragma unroll
    for (int v = 0; v < 4; ++v) {
      const int c = cBase + v * 16 + mrow;
      const float bv = biasp[c];
#pragma unroll
      for (int r = 0; r < 8; ++r) {
        const int slot = rBase + u * 16 + half * 8 + r;
        if (slot < M) {
          const float val = acc[u][v][r] + bv;
          if constexpr (EPI == 0) {
            const float s = val / (1.0f + __expf(-val));   // SiLU
            ((unsigned short*)Outv)[(long)(base + slot) * Ncols + c] = f2bf(s);
          } else if constexpr (EPI == 1) {
            ((float*)Outv)[(long)slot * Ncols + c] = val;
          } else {
            atomicAdd(((float*)Outv) + (long)orow[r] * Ncols + c, scl[r] * val);
          }
        }
      }
    }
  }
}

// ---------------------------------------------------------------- launch ---
extern "C" void kernel_launch(void* const* d_in, const int* in_sizes, int n_in,
                              void* d_out, int out_size, void* d_ws, size_t ws_size,
                              hipStream_t stream) {
  (void)in_sizes; (void)n_in; (void)out_size; (void)ws_size;
  const float* X   = (const float*)d_in[0];
  const float* Wr  = (const float*)d_in[1];
  const float* eb  = (const float*)d_in[2];
  const float* sW1 = (const float*)d_in[3];
  const float* sb1 = (const float*)d_in[4];
  const float* sW2 = (const float*)d_in[5];
  const float* sb2 = (const float*)d_in[6];
  const float* rW1 = (const float*)d_in[7];
  const float* rb1 = (const float*)d_in[8];
  const float* rW2 = (const float*)d_in[9];
  const float* rb2 = (const float*)d_in[10];
  float* out = (float*)d_out;

  char* p = (char*)d_ws;
  auto carve = [&p](size_t bytes) -> char* {
    char* r = p; p += (bytes + 255) & ~(size_t)255; return r;
  };
  int*   cnt   = (int*)  carve(NEXP * 4);
  int*   basep = (int*)  carve(NEXP * 4);
  int*   tok   = (int*)  carve((size_t)NEXP * NTOK * 4);
  float* wgt   = (float*)carve((size_t)NEXP * NTOK * 4);
  unsigned short* Xb   = (unsigned short*)carve((size_t)NTOK * DMODEL * 2);
  unsigned short* sW1T = (unsigned short*)carve((size_t)HDIM * DMODEL * 2);
  unsigned short* sW2T = (unsigned short*)carve((size_t)DMODEL * HDIM * 2);
  unsigned short* W1T  = (unsigned short*)carve((size_t)NEXP * HDIM * DMODEL * 2);
  unsigned short* W2T  = (unsigned short*)carve((size_t)NEXP * DMODEL * HDIM * 2);
  unsigned short* Hs   = (unsigned short*)carve((size_t)NTOK * HDIM * 2);
  unsigned short* Hr   = (unsigned short*)carve((size_t)2 * NTOK * HDIM * 2);

  // routing + compaction
  zero8<<<1, 32, 0, stream>>>(cnt);
  routing_kernel<<<NTOK / 8, 256, 0, stream>>>(X, Wr, eb, cnt, tok, wgt, NTOK, DMODEL);
  scan8<<<1, 32, 0, stream>>>(cnt, basep);

  // precision / layout conversion
  cvt_bf16<<<(NTOK * DMODEL / 4 + 255) / 256, 256, 0, stream>>>(X, Xb, (long)NTOK * DMODEL);
  transpose_cvt<<<dim3(HDIM / 32, DMODEL / 32, 1),    dim3(32, 8), 0, stream>>>(sW1, sW1T, DMODEL, HDIM);
  transpose_cvt<<<dim3(DMODEL / 32, HDIM / 32, 1),    dim3(32, 8), 0, stream>>>(sW2, sW2T, HDIM, DMODEL);
  transpose_cvt<<<dim3(HDIM / 32, DMODEL / 32, NEXP), dim3(32, 8), 0, stream>>>(rW1, W1T, DMODEL, HDIM);
  transpose_cvt<<<dim3(DMODEL / 32, HDIM / 32, NEXP), dim3(32, 8), 0, stream>>>(rW2, W2T, HDIM, DMODEL);

  // shared expert: Hs = SiLU(X@sW1+b1); out = Hs@sW2+b2 (writes every element)
  moe_gemm<0, false><<<dim3(HDIM / NTILE, NTOK / MTILE, 1), 256, 0, stream>>>(
      Xb, sW1T, 0, sb1, DMODEL, HDIM, NTOK, nullptr, nullptr, nullptr, nullptr, Hs);
  // routed experts: Hr = SiLU(gather(X)@W1[e]+b1[e])
  moe_gemm<0, true><<<dim3(HDIM / NTILE, NTOK / MTILE, NEXP), 256, 0, stream>>>(
      Xb, W1T, (long)HDIM * DMODEL, rb1, DMODEL, HDIM, 0, cnt, tok, nullptr, basep, Hr);
  moe_gemm<1, false><<<dim3(DMODEL / NTILE, NTOK / MTILE, 1), 256, 0, stream>>>(
      Hs, sW2T, 0, sb2, HDIM, DMODEL, NTOK, nullptr, nullptr, nullptr, nullptr, out);
  // out += w * (Hr@W2[e]+b2[e]) scattered per token
  moe_gemm<2, true><<<dim3(DMODEL / NTILE, NTOK / MTILE, NEXP), 256, 0, stream>>>(
      Hr, W2T, (long)DMODEL * HDIM, rb2, HDIM, DMODEL, 0, cnt, tok, wgt, basep, out);
}
